// TiNeuVox_4045859193530
// MI455X (gfx1250) — compile-verified, matrix operand order checked by reading, not run on hardware
//
#include <hip/hip_runtime.h>
#include <hip/hip_bf16.h>
#include <math.h>

// ---------------------------------------------------------------------------
// TiNeuVox forward for MI455X (gfx1250, wave32, WMMA).
// All big MLP layers run as f16 WMMA GEMMs (v_wmma_f32_16x16x32_f16) with
// f32 accumulation; activations are f16 to halve HBM traffic (traffic-bound:
// ~0.9GB RW @ 23.3TB/s). Weights pre-packed into the ISA 32x16 B-fragment
// lane layout so each lane loads its 16 halves with one 32B contiguous load.
// ---------------------------------------------------------------------------

typedef __attribute__((ext_vector_type(16))) _Float16 v16h;
typedef __attribute__((ext_vector_type(8)))  _Float16 v8h;
typedef __attribute__((ext_vector_type(8)))  float    v8f;

#define NRAYS   2048
#define NSAMP   128
#define NPTS    (NRAYS * NSAMP)          // 262144
#define ACT_SHIFT (-4.595119850134589f)  // log(1/(1-0.01)-1)
#define INTERVAL  0.5f

// ============================ weight pre-pack ==============================
// Packed layout per (ktile,ntile): 32 lanes x 16 halves, half h of lane L is
// element K = (h<8 ? h+8g : h+8+8g), N = L&15, g = L>>4 (mirrors the ISA
// 16-bit A-matrix 16x32 K mapping, applied to the B operand with N=lane%16).
__global__ __launch_bounds__(256) void k_pack(const float* __restrict__ W,
                                              _Float16* __restrict__ Bp,
                                              int din, int dout,
                                              int ktiles, int ntiles) {
  int idx = blockIdx.x * 256 + threadIdx.x;
  int total = ktiles * ntiles * 512;
  if (idx >= total) return;
  int h    = idx & 15;
  int lane = (idx >> 4) & 31;
  int tile = idx >> 9;
  int kt = tile / ntiles, nt = tile - kt * ntiles;
  int g  = lane >> 4;
  int kk = (h < 8) ? (h + 8 * g) : (h + 8 + 8 * g);
  int k  = kt * 32 + kk;
  int n  = nt * 16 + (lane & 15);
  float v = (k < din && n < dout) ? W[(size_t)k * dout + n] : 0.0f;
  Bp[idx] = (_Float16)v;
}

// ============================ per-ray MLP ==================================
__global__ __launch_bounds__(256) void k_ray(const float* __restrict__ viewdirs,
                                             const float* __restrict__ times,
                                             const float* __restrict__ t1w,
                                             const float* __restrict__ t1b,
                                             const float* __restrict__ t2w,
                                             const float* __restrict__ t2b,
                                             float* __restrict__ t_feat,
                                             float* __restrict__ vde) {
  __shared__ float te[17];
  __shared__ float hid[256];
  int ray = blockIdx.x;
  int tid = threadIdx.x;
  float t = times[ray];
  if (tid == 0) te[0] = t;
  if (tid >= 1 && tid < 9)  te[tid] = sinf(t * (float)(1 << (tid - 1)));
  if (tid >= 9 && tid < 17) te[tid] = cosf(t * (float)(1 << (tid - 9)));
  __syncthreads();
  float acc = t1b[tid];
  for (int j = 0; j < 17; j++) acc = fmaf(te[j], t1w[j * 256 + tid], acc);
  hid[tid] = acc > 0.f ? acc : 0.f;
  __syncthreads();
  if (tid < 30) {
    float a2 = t2b[tid];
    for (int j = 0; j < 256; j++) a2 = fmaf(hid[j], t2w[j * 30 + tid], a2);
    t_feat[ray * 30 + tid] = a2;
  }
  if (tid < 27) {
    float d[3] = { viewdirs[ray*3+0], viewdirs[ray*3+1], viewdirs[ray*3+2] };
    float v;
    if (tid < 3)        v = d[tid];
    else if (tid < 15)  { int q = tid - 3;  v = sinf(d[q >> 2] * (float)(1 << (q & 3))); }
    else                { int q = tid - 15; v = cosf(d[q >> 2] * (float)(1 << (q & 3))); }
    vde[ray * 27 + tid] = v;
  }
}

// Fold viewdir part of rgb_v1 into a per-ray bias: vbias = b1 + vde @ W1[256:283]
__global__ __launch_bounds__(256) void k_vbias(const float* __restrict__ vde,
                                               const float* __restrict__ w1,
                                               const float* __restrict__ b1,
                                               float* __restrict__ vbias) {
  int i = blockIdx.x * 256 + threadIdx.x;
  if (i >= NRAYS * 128) return;
  int ray = i >> 7, n = i & 127;
  float a = b1[n];
  for (int j = 0; j < 27; j++)
    a = fmaf(vde[ray * 27 + j], w1[(size_t)(256 + j) * 128 + n], a);
  vbias[i] = a;
}

// ===================== per-point setup (pts_emb, concat) ===================
__global__ __launch_bounds__(256) void k_setup(const float* __restrict__ rays_o,
                                               const float* __restrict__ rays_d,
                                               const float* __restrict__ t_feat,
                                               _Float16* __restrict__ A0,   // P x 96
                                               _Float16* __restrict__ A1,   // P x 192
                                               float* __restrict__ ray_pts) {
  int p = blockIdx.x * 256 + threadIdx.x;
  if (p >= NPTS) return;
  int ray = p >> 7, s = p & 127;
  float t = 0.2f + 2.3f * ((float)s + 0.5f) / 128.0f;
  float pt[3];
  #pragma unroll
  for (int c = 0; c < 3; c++) {
    pt[c] = rays_o[ray * 3 + c] + rays_d[ray * 3 + c] * t;
    ray_pts[p * 3 + c] = pt[c];
  }
  float pe[63];
  pe[0] = pt[0]; pe[1] = pt[1]; pe[2] = pt[2];
  #pragma unroll
  for (int dim = 0; dim < 3; dim++)
    #pragma unroll
    for (int f = 0; f < 10; f++) {
      float e = pt[dim] * (float)(1 << f);
      pe[3  + dim * 10 + f] = sinf(e);
      pe[33 + dim * 10 + f] = cosf(e);
    }
  _Float16* a0 = A0 + (size_t)p * 96;
  _Float16* a1 = A1 + (size_t)p * 192 + 90;  // cols 90..191 of featurenet input
  for (int i = 0; i < 63; i++) { _Float16 h = (_Float16)pe[i]; a0[i] = h; a1[i] = h; }
  const float* tf = t_feat + ray * 30;
  for (int i = 0; i < 30; i++) { _Float16 h = (_Float16)tf[i]; a0[63 + i] = h; a1[63 + i] = h; }
  for (int i = 93; i < 96;  i++) a0[i] = (_Float16)0.f;   // K-pad 93..95
  for (int i = 93; i < 102; i++) a1[i] = (_Float16)0.f;   // cols 183..191 pad
}

// ============================ WMMA GEMM ====================================
// One wave per 16-row M-tile; A fragments cached in VGPRs (KTILES x v16h),
// sweeps all N-tiles. out = act( A(f16) x Bpack(f16) + bias ), f32 accum.
template <int KTILES>
__global__ __launch_bounds__(256) void gemm_wmma(const _Float16* __restrict__ A,
                                                 const _Float16* __restrict__ Bp,
                                                 const float* __restrict__ bias,
                                                 _Float16* __restrict__ Out,
                                                 int ntiles, int relu, int biasPerRay) {
  const int lane  = threadIdx.x & 31;
  const int mtile = blockIdx.x * 8 + (threadIdx.x >> 5);
  const int g     = lane >> 4;
  const int ld    = KTILES * 32;
  const int m     = mtile * 16 + (lane & 15);
  const _Float16* arow = A + (size_t)m * ld + g * 8;

  v16h afrag[KTILES];
  #pragma unroll
  for (int kt = 0; kt < KTILES; kt++) {
    v8h lo = *(const v8h*)(arow + kt * 32);        // K = kt*32 + 8g .. +7
    v8h hi = *(const v8h*)(arow + kt * 32 + 16);   // K = kt*32 + 16 + 8g ..
    #pragma unroll
    for (int i = 0; i < 8; i++) { afrag[kt][i] = lo[i]; afrag[kt][i + 8] = hi[i]; }
  }

  const int N = ntiles * 16;
  for (int nt = 0; nt < ntiles; nt++) {
    float bv = biasPerRay
             ? bias[(size_t)(mtile >> 3) * N + nt * 16 + (lane & 15)]  // 16 rows/tile, 128 samples/ray
             : bias[nt * 16 + (lane & 15)];
    v8f acc;
    #pragma unroll
    for (int r = 0; r < 8; r++) acc[r] = bv;
    #pragma unroll
    for (int kt = 0; kt < KTILES; kt++) {
      const _Float16* bp = Bp + ((size_t)(kt * ntiles + nt) * 512) + lane * 16;
      v16h bfrag = *(const v16h*)bp;  // one contiguous 32B load per lane
      acc = __builtin_amdgcn_wmma_f32_16x16x32_f16(
          false, afrag[kt], false, bfrag, (short)0, acc, false, false);
    }
    #pragma unroll
    for (int r = 0; r < 8; r++) {
      float v = acc[r];
      if (relu) v = v > 0.f ? v : 0.f;
      int row = mtile * 16 + r + 8 * g;            // C/D layout: M = r + 8*(lane>>4)
      Out[(size_t)row * N + nt * 16 + (lane & 15)] = (_Float16)v;
    }
  }
}

// ==================== def_out + deformation + grid sample ==================
__device__ inline void sample_scale(const float* __restrict__ G,
                                    float px, float py, float pz,
                                    int s, float* out6) {
  float fx = floorf(px), fy = floorf(py), fz = floorf(pz);
  int ix = (int)fx, iy = (int)fy, iz = (int)fz;
  float tx = px - fx, ty = py - fy, tz = pz - fz;
  #pragma unroll
  for (int d = 0; d < 8; d++) {
    int ox = (d >> 2) & 1, oy = (d >> 1) & 1, oz = d & 1;
    int jx = ix + ox, jy = iy + oy, jz = iz + oz;
    float w = (ox ? tx : 1.f - tx) * (oy ? ty : 1.f - ty) * (oz ? tz : 1.f - tz);
    // padded grid: idx*s == 160 hits the zero pad; OOB masked -> contribute 0
    bool ok = (jx >= 0) && (jy >= 0) && (jz >= 0) &&
              (jx * s < 160) && (jy * s < 160) && (jz * s < 160);
    if (ok) {
      size_t base = (size_t)(jx * s) * 25600 + (size_t)(jy * s) * 160 + (size_t)(jz * s);
      #pragma unroll
      for (int c = 0; c < 6; c++) out6[c] += w * G[(size_t)c * 4096000 + base];
    }
  }
}

__global__ __launch_bounds__(256) void k_defout_sample(const _Float16* __restrict__ H2,
                                                       const float* __restrict__ wout,
                                                       const float* __restrict__ bout,
                                                       const float* __restrict__ ray_pts,
                                                       const float* __restrict__ G,
                                                       _Float16* __restrict__ A1) {
  int p = blockIdx.x * 256 + threadIdx.x;
  if (p >= NPTS) return;
  const _Float16* h = H2 + (size_t)p * 256;
  float d0 = bout[0], d1 = bout[1], d2 = bout[2];
  for (int j = 0; j < 256; j++) {
    float hv = (float)h[j];
    d0 = fmaf(hv, wout[j * 3 + 0], d0);
    d1 = fmaf(hv, wout[j * 3 + 1], d1);
    d2 = fmaf(hv, wout[j * 3 + 2], d2);
  }
  float u = (ray_pts[p * 3 + 0] + d0 + 1.5f) / 3.0f;
  float v = (ray_pts[p * 3 + 1] + d1 + 1.5f) / 3.0f;
  float w = (ray_pts[p * 3 + 2] + d2 + 1.5f) / 3.0f;
  float vox[18];
  #pragma unroll
  for (int i = 0; i < 18; i++) vox[i] = 0.f;
  sample_scale(G, u * 160.f, v * 160.f, w * 160.f, 1, vox + 0);
  sample_scale(G, u * 80.f,  v * 80.f,  w * 80.f,  2, vox + 6);
  sample_scale(G, u * 40.f,  v * 40.f,  w * 40.f,  4, vox + 12);
  _Float16* a1 = A1 + (size_t)p * 192;       // cols 0..89 = vox_emb
  #pragma unroll
  for (int i = 0; i < 18; i++) a1[i] = (_Float16)vox[i];
  #pragma unroll
  for (int i = 0; i < 18; i++) {
    a1[18 + i * 2 + 0] = (_Float16)sinf(vox[i]);
    a1[18 + i * 2 + 1] = (_Float16)sinf(vox[i] * 2.f);
    a1[54 + i * 2 + 0] = (_Float16)cosf(vox[i]);
    a1[54 + i * 2 + 1] = (_Float16)cosf(vox[i] * 2.f);
  }
}

// ============================ density / alpha ==============================
__global__ __launch_bounds__(256) void k_density(const _Float16* __restrict__ HF,
                                                 const float* __restrict__ wd,
                                                 const float* __restrict__ bd,
                                                 float* __restrict__ alpha) {
  int p = blockIdx.x * 256 + threadIdx.x;
  if (p >= NPTS) return;
  const _Float16* h = HF + (size_t)p * 256;
  float a = bd[0];
  for (int j = 0; j < 256; j++) a = fmaf((float)h[j], wd[j], a);
  float x  = a + ACT_SHIFT;
  float sp = x > 20.f ? x : log1pf(expf(x));     // softplus
  alpha[p] = 1.f - expf(-sp * INTERVAL);
}

// ============================ rgb head (128->3) ============================
__global__ __launch_bounds__(256) void k_rgb(const _Float16* __restrict__ RH,
                                             const float* __restrict__ w2,
                                             const float* __restrict__ b2,
                                             float* __restrict__ rgb) {
  int p = blockIdx.x * 256 + threadIdx.x;
  if (p >= NPTS) return;
  const _Float16* h = RH + (size_t)p * 128;
  float r = b2[0], g = b2[1], b = b2[2];
  for (int j = 0; j < 128; j++) {
    float hv = (float)h[j];
    r = fmaf(hv, w2[j * 3 + 0], r);
    g = fmaf(hv, w2[j * 3 + 1], g);
    b = fmaf(hv, w2[j * 3 + 2], b);
  }
  rgb[p * 3 + 0] = 1.f / (1.f + expf(-r));
  rgb[p * 3 + 1] = 1.f / (1.f + expf(-g));
  rgb[p * 3 + 2] = 1.f / (1.f + expf(-b));
}

// ============================ ray march scan ===============================
__global__ __launch_bounds__(256) void k_march(const float* __restrict__ alpha,
                                               const float* __restrict__ rgb,
                                               float* __restrict__ out) {
  int ray = blockIdx.x * 256 + threadIdx.x;
  if (ray >= NRAYS) return;
  float T = 1.f, r = 0.f, g = 0.f, b = 0.f;
  for (int s = 0; s < NSAMP; s++) {
    int p = ray * NSAMP + s;
    float a = alpha[p];
    float w = a * T;
    r += w * rgb[p * 3 + 0];
    g += w * rgb[p * 3 + 1];
    b += w * rgb[p * 3 + 2];
    T *= (1.f - a);
  }
  out[ray * 3 + 0] = r + T;   // BG = 1.0
  out[ray * 3 + 1] = g + T;
  out[ray * 3 + 2] = b + T;
}

// ============================ host launcher ================================
extern "C" void kernel_launch(void* const* d_in, const int* in_sizes, int n_in,
                              void* d_out, int out_size, void* d_ws, size_t ws_size,
                              hipStream_t stream) {
  (void)in_sizes; (void)n_in; (void)out_size; (void)ws_size;
  // setup_inputs order: rays_o, rays_d, viewdirs, times_sel, then params
  // (dict insertion order, each linear = {w,b}), feature grid last.
  const float* rays_o   = (const float*)d_in[0];
  const float* rays_d   = (const float*)d_in[1];
  const float* viewdirs = (const float*)d_in[2];
  const float* times    = (const float*)d_in[3];
  const float* t1w = (const float*)d_in[4],  *t1b = (const float*)d_in[5];
  const float* t2w = (const float*)d_in[6],  *t2b = (const float*)d_in[7];
  const float* d0w = (const float*)d_in[8],  *d0b = (const float*)d_in[9];
  const float* d1w = (const float*)d_in[10], *d1b = (const float*)d_in[11];
  const float* dow = (const float*)d_in[12], *dob = (const float*)d_in[13];
  const float* fnw = (const float*)d_in[14], *fnb = (const float*)d_in[15];
  const float* dnw = (const float*)d_in[16], *dnb = (const float*)d_in[17];
  const float* rfw = (const float*)d_in[18], *rfb = (const float*)d_in[19];
  const float* r1w = (const float*)d_in[20], *r1b = (const float*)d_in[21];
  const float* r2w = (const float*)d_in[22], *r2b = (const float*)d_in[23];
  const float* grid = (const float*)d_in[24];
  float* out = (float*)d_out;

  char* ws = (char*)d_ws;
  size_t off = 0;
  auto alloc = [&](size_t bytes) -> char* {
    char* p = ws + off;
    off += (bytes + 255) & ~(size_t)255;
    return p;
  };
  _Float16* bufB   = (_Float16*)alloc((size_t)NPTS * 256 * 2); // H1 / HF / RH
  _Float16* bufC   = (_Float16*)alloc((size_t)NPTS * 256 * 2); // A0 / H2 / FR
  _Float16* bufD   = (_Float16*)alloc((size_t)NPTS * 192 * 2); // featurenet input
  float* ray_pts   = (float*)alloc((size_t)NPTS * 3 * 4);
  float* t_feat    = (float*)alloc((size_t)NRAYS * 30 * 4);
  float* vde       = (float*)alloc((size_t)NRAYS * 27 * 4);
  float* vbias     = (float*)alloc((size_t)NRAYS * 128 * 4);
  float* alpha     = (float*)alloc((size_t)NPTS * 4);
  float* rgbb      = (float*)alloc((size_t)NPTS * 3 * 4);
  _Float16* pk_d0  = (_Float16*)alloc((size_t)3 * 16 * 512 * 2);
  _Float16* pk_d1  = (_Float16*)alloc((size_t)8 * 16 * 512 * 2);
  _Float16* pk_fn  = (_Float16*)alloc((size_t)6 * 16 * 512 * 2);
  _Float16* pk_rf  = (_Float16*)alloc((size_t)8 * 16 * 512 * 2);
  _Float16* pk_r1  = (_Float16*)alloc((size_t)8 * 8  * 512 * 2);

  // ---- weight packing (tiny) ----
  k_pack<<<(3*16*512 + 255)/256, 256, 0, stream>>>(d0w, pk_d0,  93, 256, 3, 16);
  k_pack<<<(8*16*512 + 255)/256, 256, 0, stream>>>(d1w, pk_d1, 256, 256, 8, 16);
  k_pack<<<(6*16*512 + 255)/256, 256, 0, stream>>>(fnw, pk_fn, 183, 256, 6, 16);
  k_pack<<<(8*16*512 + 255)/256, 256, 0, stream>>>(rfw, pk_rf, 256, 256, 8, 16);
  k_pack<<<(8*8*512  + 255)/256, 256, 0, stream>>>(r1w, pk_r1, 256, 128, 8, 8);

  // ---- per-ray nets and embeddings ----
  k_ray<<<NRAYS, 256, 0, stream>>>(viewdirs, times, t1w, t1b, t2w, t2b, t_feat, vde);
  k_vbias<<<(NRAYS*128)/256, 256, 0, stream>>>(vde, r1w, r1b, vbias);

  // ---- point setup: A0 = [pts_emb | tf | pad], A1 cols 90..191 ----
  k_setup<<<NPTS/256, 256, 0, stream>>>(rays_o, rays_d, t_feat, bufC, bufD, ray_pts);

  const int MBLK = (NPTS / 16) / 8;  // 2048 blocks, 8 waves (M-tiles) per block
  // def0: 96->256 relu
  gemm_wmma<3><<<MBLK, 256, 0, stream>>>(bufC, pk_d0, d0b, bufB, 16, 1, 0);
  // def1: 256->256 relu
  gemm_wmma<8><<<MBLK, 256, 0, stream>>>(bufB, pk_d1, d1b, bufC, 16, 1, 0);
  // def_out + deform + 3-scale grid sample -> A1 cols 0..89
  k_defout_sample<<<NPTS/256, 256, 0, stream>>>(bufC, dow, dob, ray_pts, grid, bufD);
  // featurenet: 192->256 relu
  gemm_wmma<6><<<MBLK, 256, 0, stream>>>(bufD, pk_fn, fnb, bufB, 16, 1, 0);
  // density -> alpha
  k_density<<<NPTS/256, 256, 0, stream>>>(bufB, dnw, dnb, alpha);
  // rgb_feat: 256->256 (no act)
  gemm_wmma<8><<<MBLK, 256, 0, stream>>>(bufB, pk_rf, rfb, bufC, 16, 0, 0);
  // rgb_v1: 256->128 relu, per-ray bias carries the viewdir contribution
  gemm_wmma<8><<<MBLK, 256, 0, stream>>>(bufC, pk_r1, vbias, bufB, 8, 1, 1);
  // rgb_v2 + sigmoid
  k_rgb<<<NPTS/256, 256, 0, stream>>>(bufB, r2w, r2b, rgbb);
  // ray march
  k_march<<<NRAYS/256, 256, 0, stream>>>(alpha, rgbb, out);
}
